// Encoder_78340203479339
// MI455X (gfx1250) — compile-verified, matrix operand order checked by reading, not run on hardware
//
#include <hip/hip_runtime.h>

typedef unsigned short u16;
typedef __attribute__((ext_vector_type(16))) __bf16 v16bf;
typedef __attribute__((ext_vector_type(8)))  float  v8f;

union AF { v16bf v; unsigned u[8]; };

#define BDIM 256

__device__ __forceinline__ u16 f2bf(float f) {
  unsigned u = __float_as_uint(f);
  unsigned r = u + 0x7FFFu + ((u >> 16) & 1u);   // round-to-nearest-even
  return (u16)(r >> 16);
}
__device__ __forceinline__ float sigm(float x) { return 1.0f / (1.0f + __expf(-x)); }

// ---------------------------------------------------------------------------
// Generic bf16 WMMA GEMM:  C[M,N] = scale * (A[M,K] * Bw[N,K]^T) + bias[N]
// Requires M%128==0, N%128==0, K%32==0. Optional bf16 mirror of C.
// Staging: CDNA5 async memory->LDS DMA (global_load_async_to_lds_b128),
// double-buffered LDS tiles, ASYNCcnt-based pipelining.
// ---------------------------------------------------------------------------
__global__ __launch_bounds__(256) void enc_gemm_bf16_nt(
    const u16* __restrict__ A, const u16* __restrict__ Bw,
    const float* __restrict__ bias, float* __restrict__ C, u16* __restrict__ Cbf,
    int M, int N, int K, float scale)
{
  __shared__ u16 shA[2][128 * 40];   // 32-col K slab, rows padded to 40 bf16
  __shared__ u16 shB[2][128 * 40];
  const int tid  = threadIdx.x;
  const int lane = tid & 31;
  const int wave = tid >> 5;
  const int wm = wave >> 1, wn = wave & 1;     // 4x2 wave grid -> 32x64 C per wave
  const int l15 = lane & 15, half = lane >> 4;
  const int bm = blockIdx.y * 128, bn = blockIdx.x * 128;
  const u16* Ag = A + (size_t)bm * K;
  const u16* Bg = Bw + (size_t)bn * K;

  v8f acc[2][4];
#pragma unroll
  for (int i = 0; i < 2; ++i)
#pragma unroll
    for (int j = 0; j < 4; ++j)
#pragma unroll
      for (int e = 0; e < 8; ++e) acc[i][j][e] = 0.0f;

  // Per-thread 16B chunk assignment: chunk0 = row r0, chunk1 = row r0+64.
  const int r0 = tid >> 2;           // 0..63
  const int c0 = (tid & 3) * 8;      // bf16 element offset within 32-wide slab
  const u16* gA0 = Ag + (size_t)r0 * K + c0;
  const u16* gA1 = Ag + (size_t)(r0 + 64) * K + c0;
  const u16* gB0 = Bg + (size_t)r0 * K + c0;
  const u16* gB1 = Bg + (size_t)(r0 + 64) * K + c0;
  const unsigned dA0 = (unsigned)(size_t)&shA[0][r0 * 40 + c0];
  const unsigned dA1 = (unsigned)(size_t)&shA[0][(r0 + 64) * 40 + c0];
  const unsigned dB0 = (unsigned)(size_t)&shB[0][r0 * 40 + c0];
  const unsigned dB1 = (unsigned)(size_t)&shB[0][(r0 + 64) * 40 + c0];
  const unsigned bufStep = (unsigned)(128 * 40 * sizeof(u16));   // 10240 B

#define ENC_ASYNC_SLAB(KT_IDX, PBUF)                                           \
  {                                                                            \
    const unsigned d_ = (PBUF) ? bufStep : 0u;                                 \
    const size_t ko_ = (size_t)(KT_IDX) * 32;                                  \
    asm volatile("global_load_async_to_lds_b128 %0, %1, off"                   \
                 :: "v"(dA0 + d_), "v"(gA0 + ko_) : "memory");                 \
    asm volatile("global_load_async_to_lds_b128 %0, %1, off"                   \
                 :: "v"(dA1 + d_), "v"(gA1 + ko_) : "memory");                 \
    asm volatile("global_load_async_to_lds_b128 %0, %1, off"                   \
                 :: "v"(dB0 + d_), "v"(gB0 + ko_) : "memory");                 \
    asm volatile("global_load_async_to_lds_b128 %0, %1, off"                   \
                 :: "v"(dB1 + d_), "v"(gB1 + ko_) : "memory");                 \
  }

  ENC_ASYNC_SLAB(0, 0);

  const int KT = K >> 5;
#pragma unroll 2
  for (int kt = 0; kt < KT; ++kt) {
    const int p = kt & 1;
    if (kt + 1 < KT) {
      ENC_ASYNC_SLAB(kt + 1, p ^ 1);
      // 4 new ops in flight; wait for the previous slab's 4 to complete.
      asm volatile("s_wait_asynccnt 0x4" ::: "memory");
    } else {
      asm volatile("s_wait_asynccnt 0x0" ::: "memory");
    }
    __syncthreads();

    AF af[2], bfr[4];
    const u16* sAp = shA[p];
    const u16* sBp = shB[p];
#pragma unroll
    for (int mt = 0; mt < 2; ++mt) {
      const u16* base = sAp + (wm * 32 + mt * 16 + l15) * 40 + half * 8;
#pragma unroll
      for (int v = 0; v < 8; ++v) {
        const int k = ((v >> 2) << 4) + ((v & 3) << 1);
        af[mt].u[v] = *(const unsigned*)(base + k);
      }
    }
#pragma unroll
    for (int nt = 0; nt < 4; ++nt) {
      const u16* base = sBp + (wn * 64 + nt * 16 + l15) * 40 + half * 8;
#pragma unroll
      for (int v = 0; v < 8; ++v) {
        const int k = ((v >> 2) << 4) + ((v & 3) << 1);
        bfr[nt].u[v] = *(const unsigned*)(base + k);
      }
    }
#pragma unroll
    for (int mt = 0; mt < 2; ++mt)
#pragma unroll
      for (int nt = 0; nt < 4; ++nt)
        acc[mt][nt] = __builtin_amdgcn_wmma_f32_16x16x32_bf16(
            false, af[mt].v, false, bfr[nt].v, (short)0, acc[mt][nt], false, false);
    __syncthreads();
  }
#undef ENC_ASYNC_SLAB

#pragma unroll
  for (int mt = 0; mt < 2; ++mt) {
#pragma unroll
    for (int nt = 0; nt < 4; ++nt) {
      const int col = bn + wn * 64 + nt * 16 + l15;
      const float bv = bias ? bias[col] : 0.0f;
#pragma unroll
      for (int v = 0; v < 8; ++v) {
        const int row = bm + wm * 32 + mt * 16 + v + 8 * half;
        const float val = acc[mt][nt][v] * scale + bv;
        const size_t o = (size_t)row * N + col;
        C[o] = val;
        if (Cbf) Cbf[o] = f2bf(val);
      }
    }
  }
}

// ---------------------------------------------------------------------------
// Elementwise / small kernels
// ---------------------------------------------------------------------------
__global__ void enc_cvt_bf16(const float* __restrict__ in, u16* __restrict__ out, int n) {
  int i = blockIdx.x * BDIM + threadIdx.x;
  if (i < n) out[i] = f2bf(in[i]);
}

// emb[:1000] -> bf16, rows 1000..1023 zero-padded (for 1024-row GEMM)
__global__ void enc_embpad(const float* __restrict__ emb, u16* __restrict__ out) {
  int i = blockIdx.x * BDIM + threadIdx.x;
  if (i >= 1024 * 512) return;
  int row = i >> 9;
  out[i] = (row < 1000) ? f2bf(emb[i]) : (u16)0;
}

__global__ void enc_gather_tokens(const int* __restrict__ tokens, const float* __restrict__ emb,
                                  float* __restrict__ X, u16* __restrict__ Xbf) {
  int i = blockIdx.x * BDIM + threadIdx.x;
  if (i >= 128 * 16 * 512) return;
  int e = i & 511, bs = i >> 9;
  float v = emb[(size_t)tokens[bs] * 512 + e];
  X[i] = v; Xbf[i] = f2bf(v);
}

// h1 = GRU(x, h=0): gh == b_hh.  GI rows are (b*L + j), j in [0, L-1).
__global__ void enc_h1(const float* __restrict__ GI, const float* __restrict__ bhh,
                       float* __restrict__ h1, u16* __restrict__ h1bf, int L) {
  int i = blockIdx.x * BDIM + threadIdx.x;
  int Lm1 = L - 1, total = 128 * Lm1 * 512;
  if (i >= total) return;
  int e = i & 511, rj = i >> 9;
  int j = rj % Lm1, b = rj / Lm1;
  const float* gi = GI + (size_t)(b * L + j) * 1536;
  float r = sigm(gi[e] + bhh[e]);
  float z = sigm(gi[512 + e] + bhh[512 + e]);
  float n = tanhf(gi[1024 + e] + r * bhh[1024 + e]);
  float h = (1.0f - z) * n;
  h1[i] = h; h1bf[i] = f2bf(h);
}

// pairs = GRU(x[:,1:], h1).  GI rows (b*L + j + 1), GH2/h1 rows (b*(L-1)+j).
__global__ void enc_pairs(const float* __restrict__ GI, const float* __restrict__ GH2,
                          const float* __restrict__ h1, float* __restrict__ pairs,
                          u16* __restrict__ pairsbf, int L) {
  int i = blockIdx.x * BDIM + threadIdx.x;
  int Lm1 = L - 1, total = 128 * Lm1 * 512;
  if (i >= total) return;
  int e = i & 511, rj = i >> 9;
  int j = rj % Lm1, b = rj / Lm1;
  const float* gi = GI + (size_t)(b * L + j + 1) * 1536;
  const float* gh = GH2 + (size_t)rj * 1536;
  float r = sigm(gi[e] + gh[e]);
  float z = sigm(gi[512 + e] + gh[512 + e]);
  float n = tanhf(gi[1024 + e] + r * gh[1024 + e]);
  float out = (1.0f - z) * n + z * h1[i];
  pairs[i] = out; pairsbf[i] = f2bf(out);
}

// logit[r] = pairs[r]·w_fc + b_fc  (sigmoid is monotone; argmax on logits)
__global__ void enc_pairscore(const float* __restrict__ pairs, const float* __restrict__ wfc,
                              const float* __restrict__ bfc, float* __restrict__ logit) {
  __shared__ float sb[128];
  int r = blockIdx.x, tid = threadIdx.x;
  float s = 0.0f;
  for (int e = tid; e < 512; e += 128) s += pairs[(size_t)r * 512 + e] * wfc[e];
  sb[tid] = s; __syncthreads();
  for (int k = 64; k > 0; k >>= 1) { if (tid < k) sb[tid] += sb[tid + k]; __syncthreads(); }
  if (tid == 0) logit[r] = sb[0] + bfc[0];
}

__global__ void enc_argmax(const float* __restrict__ logit, int* __restrict__ sel, int Lm1) {
  int b = threadIdx.x;
  if (b >= 128) return;
  const float* row = logit + b * Lm1;
  int best = 0; float bv = row[0];
  for (int j = 1; j < Lm1; ++j) { float v = row[j]; if (v > bv) { bv = v; best = j; } }
  sel[b] = best;
}

__global__ void enc_gatherpair(const float* __restrict__ pairs, const u16* __restrict__ pairsbf,
                               const int* __restrict__ sel, int Lm1,
                               float* __restrict__ pairf, u16* __restrict__ pairbf) {
  int b = blockIdx.x;
  int s = sel ? sel[b] : 0;
  const float* src = pairs + (size_t)(b * Lm1 + s) * 512;
  const u16* srcb = pairsbf + (size_t)(b * Lm1 + s) * 512;
  for (int e = threadIdx.x; e < 512; e += blockDim.x) {
    pairf[b * 512 + e] = src[e];
    pairbf[b * 512 + e] = srcb[e];
  }
}

// scores[b, 1000] = scale * q[b]·kp[b]   (scores row stride 1024)
__global__ void enc_qdotkp(const float* __restrict__ q, const float* __restrict__ kp,
                           float* __restrict__ scores, float scale) {
  __shared__ float sb[128];
  int b = blockIdx.x, tid = threadIdx.x;
  float s = 0.0f;
  for (int e = tid; e < 512; e += 128) s += q[b * 512 + e] * kp[b * 512 + e];
  sb[tid] = s; __syncthreads();
  for (int k = 64; k > 0; k >>= 1) { if (tid < k) sb[tid] += sb[tid + k]; __syncthreads(); }
  if (tid == 0) scores[(size_t)b * 1024 + 1000] = sb[0] * scale;
}

// softmax + entropy over 1001 columns (row stride 1024); optional prob output
__global__ void enc_softmax_entropy(const float* __restrict__ scores, float* __restrict__ prob,
                                    float* __restrict__ losses, int col) {
  __shared__ float sb[256];
  int b = blockIdx.x, tid = threadIdx.x;
  const float* row = scores + (size_t)b * 1024;
  float m = -3.0e38f;
  for (int i = tid; i < 1001; i += 256) m = fmaxf(m, row[i]);
  sb[tid] = m; __syncthreads();
  for (int k = 128; k > 0; k >>= 1) { if (tid < k) sb[tid] = fmaxf(sb[tid], sb[tid + k]); __syncthreads(); }
  m = sb[0]; __syncthreads();
  float Z = 0.0f, W = 0.0f;
  for (int i = tid; i < 1001; i += 256) {
    float t = row[i] - m, e = __expf(t);
    Z += e; W += e * t;
  }
  sb[tid] = Z; __syncthreads();
  for (int k = 128; k > 0; k >>= 1) { if (tid < k) sb[tid] += sb[tid + k]; __syncthreads(); }
  Z = sb[0]; __syncthreads();
  sb[tid] = W; __syncthreads();
  for (int k = 128; k > 0; k >>= 1) { if (tid < k) sb[tid] += sb[tid + k]; __syncthreads(); }
  W = sb[0]; __syncthreads();
  if (prob) {
    float inv = 1.0f / Z;
    for (int i = tid; i < 1001; i += 256) prob[(size_t)b * 1024 + i] = __expf(row[i] - m) * inv;
  }
  if (tid == 0 && losses) losses[b * 16 + col] = __logf(Z) - W / Z;
}

// merged[b] = sum_i<1000 prob[b,i]*emb[i] + prob[b,1000]*pair[b]
__global__ void enc_merged(const float* __restrict__ prob, const float* __restrict__ emb,
                           const float* __restrict__ pairf, float* __restrict__ merged) {
  __shared__ float sp[1008];
  int b = blockIdx.x, tid = threadIdx.x;
  for (int i = tid; i < 1001; i += 256) sp[i] = prob[(size_t)b * 1024 + i];
  __syncthreads();
  for (int e = tid; e < 512; e += 256) {
    float acc = sp[1000] * pairf[b * 512 + e];
    for (int i = 0; i < 1000; ++i) acc += sp[i] * emb[(size_t)i * 512 + e];
    merged[b * 512 + e] = acc;
  }
}

// out[b,j] = (j==sel) ? merged : X[b, j + (j>=sel+1)]
__global__ void enc_compact(const float* __restrict__ X, const float* __restrict__ merged,
                            const int* __restrict__ sel, float* __restrict__ Xn,
                            u16* __restrict__ Xnbf, int L) {
  int i = blockIdx.x * BDIM + threadIdx.x;
  int Lm1 = L - 1, total = 128 * Lm1 * 512;
  if (i >= total) return;
  int e = i & 511, rj = i >> 9;
  int j = rj % Lm1, b = rj / Lm1;
  int s = sel[b];
  float v;
  if (j == s) v = merged[b * 512 + e];
  else {
    int src = j + (j >= s + 1 ? 1 : 0);
    v = X[(size_t)(b * L + src) * 512 + e];
  }
  Xn[i] = v; Xnbf[i] = f2bf(v);
}

__global__ void enc_zloss(float* __restrict__ losses, int col) {
  int b = threadIdx.x;
  if (b < 128) losses[b * 16 + col] = 0.0f;
}

__global__ void enc_copy_scores(const float* __restrict__ scores, float* __restrict__ out) {
  int b = blockIdx.x;
  for (int i = threadIdx.x; i < 1001; i += 256)
    out[(size_t)b * 1001 + i] = scores[(size_t)b * 1024 + i];
}

// ---------------------------------------------------------------------------
extern "C" void kernel_launch(void* const* d_in, const int* in_sizes, int n_in,
                              void* d_out, int out_size, void* d_ws, size_t ws_size,
                              hipStream_t stream) {
  (void)in_sizes; (void)n_in; (void)out_size; (void)ws_size;
  const int*   tokens = (const int*)d_in[0];
  const float* emb    = (const float*)d_in[1];
  const float* W_ih   = (const float*)d_in[2];
  const float* W_hh   = (const float*)d_in[3];
  const float* b_ih   = (const float*)d_in[4];
  const float* b_hh   = (const float*)d_in[5];
  const float* w_fc   = (const float*)d_in[6];
  const float* b_fc   = (const float*)d_in[7];
  const float* Wq     = (const float*)d_in[8];
  const float* bq     = (const float*)d_in[9];
  const float* Wk     = (const float*)d_in[10];
  const float* bk     = (const float*)d_in[11];

  const float SCALE = 0.044194173824159216f;  // 1/sqrt(512)
  float* out_scores = (float*)d_out;                    // (128, 1001)
  float* out_losses = (float*)d_out + 128 * 1001;       // (128, 16)

  // -------- workspace carve-out --------
  char* p = (char*)d_ws;
  auto alloc = [&](size_t bytes) -> void* {
    void* r = (void*)p;
    p += (bytes + 255) & ~(size_t)255;
    return r;
  };
  u16* Wih_bf  = (u16*)alloc(1536 * 512 * 2);
  u16* Whh_bf  = (u16*)alloc(1536 * 512 * 2);
  u16* Wq_bf   = (u16*)alloc(512 * 512 * 2);
  u16* Wk_bf   = (u16*)alloc(512 * 512 * 2);
  u16* emb_bf  = (u16*)alloc(1024 * 512 * 2);
  float* krel  = (float*)alloc(1024 * 512 * 4);
  u16* krel_bf = (u16*)alloc(1024 * 512 * 2);
  float* X0    = (float*)alloc(128 * 16 * 512 * 4);
  float* X1    = (float*)alloc(128 * 16 * 512 * 4);
  u16* X0bf    = (u16*)alloc(128 * 16 * 512 * 2);
  u16* X1bf    = (u16*)alloc(128 * 16 * 512 * 2);
  float* GI    = (float*)alloc((size_t)2048 * 1536 * 4);
  float* GH2   = (float*)alloc((size_t)1920 * 1536 * 4);
  float* h1    = (float*)alloc(128 * 15 * 512 * 4);
  u16* h1bf    = (u16*)alloc(128 * 15 * 512 * 2);
  float* pairs = (float*)alloc(128 * 15 * 512 * 4);
  u16* pairsbf = (u16*)alloc(128 * 15 * 512 * 2);
  float* logit = (float*)alloc(128 * 15 * 4);
  int* sel     = (int*)alloc(128 * 4);
  float* pairf = (float*)alloc(128 * 512 * 4);
  u16* pairbf  = (u16*)alloc(128 * 512 * 2);
  float* qf    = (float*)alloc(128 * 512 * 4);
  u16* qbf     = (u16*)alloc(128 * 512 * 2);
  float* kpf   = (float*)alloc(128 * 512 * 4);
  float* scores= (float*)alloc(128 * 1024 * 4);
  float* prob  = (float*)alloc(128 * 1024 * 4);
  float* merged= (float*)alloc(128 * 512 * 4);

  auto gemm = [&](const u16* A, const u16* B, const float* bias, float* C, u16* Cbf,
                  int M, int N, int K, float scale) {
    dim3 g(N / 128, M / 128);
    enc_gemm_bf16_nt<<<g, 256, 0, stream>>>(A, B, bias, C, Cbf, M, N, K, scale);
  };
  auto blocks = [](int total) { return (total + BDIM - 1) / BDIM; };

  // -------- one-time precompute --------
  enc_cvt_bf16<<<blocks(1536 * 512), BDIM, 0, stream>>>(W_ih, Wih_bf, 1536 * 512);
  enc_cvt_bf16<<<blocks(1536 * 512), BDIM, 0, stream>>>(W_hh, Whh_bf, 1536 * 512);
  enc_cvt_bf16<<<blocks(512 * 512), BDIM, 0, stream>>>(Wq, Wq_bf, 512 * 512);
  enc_cvt_bf16<<<blocks(512 * 512), BDIM, 0, stream>>>(Wk, Wk_bf, 512 * 512);
  enc_embpad<<<blocks(1024 * 512), BDIM, 0, stream>>>(emb, emb_bf);
  enc_gather_tokens<<<blocks(128 * 16 * 512), BDIM, 0, stream>>>(tokens, emb, X0, X0bf);
  // k_rel = emb[:1000] @ Wk^T + bk  (padded rows harmless; overwritten/unused)
  gemm(emb_bf, Wk_bf, bk, krel, krel_bf, 1024, 512, 512, 1.0f);

  float* Xc = X0;  u16* Xcbf = X0bf;
  float* Xn = X1;  u16* Xnbf = X1bf;

  // -------- 14 merge rounds: L = 16 .. 3 --------
  for (int t = 0; t < 14; ++t) {
    const int L = 16 - t, Lm1 = L - 1;
    gemm(Xcbf, Wih_bf, b_ih, GI, nullptr, 128 * L, 1536, 512, 1.0f);
    enc_h1<<<blocks(128 * Lm1 * 512), BDIM, 0, stream>>>(GI, b_hh, h1, h1bf, L);
    gemm(h1bf, Whh_bf, b_hh, GH2, nullptr, 128 * Lm1, 1536, 512, 1.0f);
    enc_pairs<<<blocks(128 * Lm1 * 512), BDIM, 0, stream>>>(GI, GH2, h1, pairs, pairsbf, L);
    enc_pairscore<<<128 * Lm1, 128, 0, stream>>>(pairs, w_fc, b_fc, logit);
    enc_argmax<<<1, 128, 0, stream>>>(logit, sel, Lm1);
    enc_gatherpair<<<128, 128, 0, stream>>>(pairs, pairsbf, sel, Lm1, pairf, pairbf);
    gemm(pairbf, Wq_bf, bq, qf, qbf, 128, 512, 512, 1.0f);
    gemm(pairbf, Wk_bf, bk, kpf, nullptr, 128, 512, 512, 1.0f);
    gemm(qbf, krel_bf, nullptr, scores, nullptr, 128, 1024, 512, SCALE);
    enc_qdotkp<<<128, 128, 0, stream>>>(qf, kpf, scores, SCALE);
    enc_softmax_entropy<<<128, 256, 0, stream>>>(scores, prob, out_losses, t);
    enc_merged<<<128, 256, 0, stream>>>(prob, emb, pairf, merged);
    enc_compact<<<blocks(128 * Lm1 * 512), BDIM, 0, stream>>>(Xc, merged, sel, Xn, Xnbf, L);
    { float* tf = Xc; Xc = Xn; Xn = tf; u16* tb = Xcbf; Xcbf = Xnbf; Xnbf = tb; }
  }

  // -------- L == 2 round: x = GRU(GRU(x0, 0) -> x1), loss col 14 = 0 --------
  gemm(Xcbf, Wih_bf, b_ih, GI, nullptr, 128 * 2, 1536, 512, 1.0f);
  enc_h1<<<blocks(128 * 512), BDIM, 0, stream>>>(GI, b_hh, h1, h1bf, 2);
  gemm(h1bf, Whh_bf, b_hh, GH2, nullptr, 128, 1536, 512, 1.0f);
  enc_pairs<<<blocks(128 * 512), BDIM, 0, stream>>>(GI, GH2, h1, pairs, pairsbf, 2);
  enc_zloss<<<1, 128, 0, stream>>>(out_losses, 14);

  // -------- final attention on x = pairs[:, 0] --------
  enc_gatherpair<<<128, 128, 0, stream>>>(pairs, pairsbf, nullptr, 1, pairf, pairbf);
  gemm(pairbf, Wq_bf, bq, qf, qbf, 128, 512, 512, 1.0f);
  gemm(pairbf, Wk_bf, bk, kpf, nullptr, 128, 512, 512, 1.0f);
  gemm(qbf, krel_bf, nullptr, scores, nullptr, 128, 1024, 512, SCALE);
  enc_qdotkp<<<128, 128, 0, stream>>>(qf, kpf, scores, SCALE);
  enc_copy_scores<<<128, 256, 0, stream>>>(scores, out_scores);
  enc_softmax_entropy<<<128, 256, 0, stream>>>(scores, nullptr, out_losses, 15);
}
